// EncoderBlock_65498251264369
// MI455X (gfx1250) — compile-verified, matrix operand order checked by reading, not run on hardware
//
#include <hip/hip_runtime.h>

typedef __attribute__((ext_vector_type(16))) _Float16 v16h;
typedef __attribute__((ext_vector_type(8)))  float    v8f;
typedef _Float16 half_t;

union Frag16 { v16h v; float4 f4[2]; };
union Half8  { float4 f4; half_t h[8]; };

#define D_MODELC 1024
#define NHEADS   16
#define DK       64
#define FFNC     4096
#define BATCH    4
#define SEQ      2048
#define TOK      (BATCH * SEQ)   // 8192

#if __has_builtin(__builtin_amdgcn_tensor_load_to_lds)
#define HAVE_TDM 1
typedef unsigned int v4u_ __attribute__((ext_vector_type(4)));
typedef int          v4i_ __attribute__((ext_vector_type(4)));
typedef int          v8i_ __attribute__((ext_vector_type(8)));
#else
#define HAVE_TDM 0
#endif

__device__ inline v8f zero8() {
    v8f z;
#pragma unroll
    for (int i = 0; i < 8; ++i) z[i] = 0.0f;
    return z;
}

__device__ inline unsigned int lds_addr_of(const void* p) {
    // generic (flat) pointer to a __shared__ object: low 32 bits are the LDS offset
    return (unsigned int)(unsigned long long)(size_t)p;
}

__device__ inline void async_load_lds_b128(unsigned int lds, const void* gaddr) {
    unsigned long long ga = (unsigned long long)(size_t)gaddr;
    asm volatile("global_load_async_to_lds_b128 %0, %1, off"
                 :: "v"(lds), "v"(ga) : "memory");
}
__device__ inline void wait_asynccnt0() {
    asm volatile("s_wait_asynccnt 0" ::: "memory");
}

#if HAVE_TDM
// TDM: DMA one contiguous 32x64-half K tile into LDS with HW padding to a 72-half
// row stride (data_size=8B, tile 16x32 units, pad 4 DWORDs every 32 DWORDs).
__device__ inline void tdm_load_k_tile(const half_t* gptr, unsigned int lds) {
    unsigned long long ga = (unsigned long long)(size_t)gptr;
    v4u_ g0;
    g0[0] = 1u;                                          // count=1
    g0[1] = lds;                                         // lds_addr
    g0[2] = (unsigned int)ga;                            // global_addr[31:0]
    g0[3] = (unsigned int)((ga >> 32) & 0x1FFFFFFull) | (2u << 30);  // [56:32], type=2
    v8i_ g1;
    g1[0] = (3 << 16) | (1 << 20) | (4 << 22) | (3 << 25); // 8B, pad_en, ivl=32DW, amt=4DW
    g1[1] = (16 << 16);                                  // tensor_dim0 = 16
    g1[2] = (32 << 16);                                  // tensor_dim1 = 32
    g1[3] = (16 << 16);                                  // tile_dim0 = 16
    g1[4] = 32;                                          // tile_dim1 = 32
    g1[5] = 16;                                          // tensor_dim0_stride = 16
    g1[6] = 0;
    g1[7] = 0;
    v4i_ gz4 = (v4i_)0;
    v8i_ gz8 = (v8i_)0;
    __builtin_amdgcn_tensor_load_to_lds(g0, g1, gz4, gz4, gz8, 0);
}
#endif

// ---------------------------------------------------------------- conversions
__global__ void k_f32_to_f16(const float* __restrict__ src, half_t* __restrict__ dst, int n) {
    int i = blockIdx.x * 256 + threadIdx.x;
    if (i < n) dst[i] = (half_t)src[i];
}

// Build fused QKV weight: out[d][c], c in [0,3072): 0..1023 Q(h*64+k), 1024.. K, 2048.. V
__global__ void k_build_wqkv(const float* __restrict__ Wq, const float* __restrict__ Wk,
                             const float* __restrict__ Wv, half_t* __restrict__ out) {
    int i = blockIdx.x * 256 + threadIdx.x;           // over 1024*3072
    int d = i / 3072, c = i % 3072;
    int sec = c >> 10, cc = c & 1023;
    int h = cc >> 6, k = cc & 63;
    const float* W = (sec == 0) ? Wq : ((sec == 1) ? Wk : Wv);
    out[i] = (half_t)W[((size_t)h * D_MODELC + d) * DK + k];
}

// ---------------------------------------------------------------- WMMA GEMM
// C[M,N](f32) and/or Ch[M,N](f16) = A[M,K]f16 * B[K,N]f16 (+bias)(relu)
// 256 threads, block tile 128x128, K-step 32, 8 waves of 32x64 wave tiles.
// Software-pipelined (double-buffered LDS): A tile goes global->LDS by async DMA
// (ASYNCcnt), B tile is transposed via VGPRs; next tile's loads overlap WMMAs.
template <bool HAS_BIAS, bool RELU, bool OUT32, bool OUT16>
__global__ void __launch_bounds__(256)
k_gemm_f16(const half_t* __restrict__ A, const half_t* __restrict__ B,
           const float* __restrict__ bias, float* __restrict__ C,
           half_t* __restrict__ Ch, int M, int N, int K) {
    __shared__ __align__(16) half_t As[2][128][40];    // row-major A tiles (pad 8)
    __shared__ __align__(16) half_t Bst[2][128][40];   // transposed B tiles: [col][k]
    int tid = threadIdx.x;
    int lane = tid & 31, wave = tid >> 5;
    int wm = wave & 3, wn = wave >> 2;            // 4 waves in M, 2 in N
    int bm = blockIdx.y * 128, bn = blockIdx.x * 128;

    v8f acc[2][4];
#pragma unroll
    for (int mt = 0; mt < 2; ++mt)
#pragma unroll
        for (int nt = 0; nt < 4; ++nt) acc[mt][nt] = zero8();

    int lrow = tid >> 1, loff = (tid & 1) * 16;   // A loader: 2 thr/row, 16 halves each
    int bc = tid & 127, bkh = (tid >> 7) * 16;    // B loader: col, k-half
    unsigned int a_lds[2] = { lds_addr_of(&As[0][lrow][loff]),
                              lds_addr_of(&As[1][lrow][loff]) };
    const half_t* Arow = A + (size_t)(bm + lrow) * K + loff;
    const half_t* Bcol = B + bn + bc;

    // prologue: tile 0 into buffer 0
    async_load_lds_b128(a_lds[0], Arow);
    async_load_lds_b128(a_lds[0] + 16, Arow + 8);
    {
        Half8 t0, t1;
#pragma unroll
        for (int i = 0; i < 8; ++i) {
            t0.h[i] = Bcol[(size_t)(bkh + i)     * N];
            t1.h[i] = Bcol[(size_t)(bkh + 8 + i) * N];
        }
        *(float4*)&Bst[0][bc][bkh]     = t0.f4;
        *(float4*)&Bst[0][bc][bkh + 8] = t1.f4;
    }
    wait_asynccnt0();
    __syncthreads();

    int buf = 0;
    for (int k0 = 0; k0 < K; k0 += 32) {
        int nxt = buf ^ 1;
        bool more = (k0 + 32) < K;
        Half8 t0, t1;
        if (more) {
            // prefetch next tile: A by async DMA, B into registers
            const half_t* ag = Arow + k0 + 32;
            async_load_lds_b128(a_lds[nxt], ag);
            async_load_lds_b128(a_lds[nxt] + 16, ag + 8);
#pragma unroll
            for (int i = 0; i < 8; ++i) {
                t0.h[i] = Bcol[(size_t)(k0 + 32 + bkh + i)     * N];
                t1.h[i] = Bcol[(size_t)(k0 + 32 + bkh + 8 + i) * N];
            }
        }

        int kb = (lane >> 4) * 8;                 // ISA 16-bit operand K packing
        Frag16 a[2], b[4];
#pragma unroll
        for (int mt = 0; mt < 2; ++mt) {
            int m = wm * 32 + mt * 16 + (lane & 15);
            a[mt].f4[0] = *(const float4*)&As[buf][m][kb];
            a[mt].f4[1] = *(const float4*)&As[buf][m][kb + 16];
        }
#pragma unroll
        for (int nt = 0; nt < 4; ++nt) {
            int c = wn * 64 + nt * 16 + (lane & 15);
            b[nt].f4[0] = *(const float4*)&Bst[buf][c][kb];
            b[nt].f4[1] = *(const float4*)&Bst[buf][c][kb + 16];
        }
#pragma unroll
        for (int mt = 0; mt < 2; ++mt)
#pragma unroll
            for (int nt = 0; nt < 4; ++nt)
                acc[mt][nt] = __builtin_amdgcn_wmma_f32_16x16x32_f16(
                    false, a[mt].v, false, b[nt].v, (short)0, acc[mt][nt], false, false);

        if (more) {
            *(float4*)&Bst[nxt][bc][bkh]     = t0.f4;
            *(float4*)&Bst[nxt][bc][bkh + 8] = t1.f4;
            wait_asynccnt0();
        }
        __syncthreads();
        buf = nxt;
    }

    int rbase = bm + wm * 32 + (lane >> 4) * 8;
    int cbase = bn + wn * 64 + (lane & 15);
#pragma unroll
    for (int mt = 0; mt < 2; ++mt)
#pragma unroll
        for (int nt = 0; nt < 4; ++nt) {
            int col = cbase + nt * 16;
            float bval = HAS_BIAS ? bias[col] : 0.0f;
            size_t base = (size_t)(rbase + mt * 16) * N + col;
#pragma unroll
            for (int j = 0; j < 8; ++j) {
                float v = acc[mt][nt][j] + bval;
                if (RELU) v = v > 0.0f ? v : 0.0f;
                if (OUT32) C[base + (size_t)j * N]  = v;
                if (OUT16) Ch[base + (size_t)j * N] = (half_t)v;
            }
        }
}

// ---------------------------------------------------------------- QKV epilogue
__global__ void k_qkv_split(const float* __restrict__ QKV,
                            const float* __restrict__ bq, const float* __restrict__ bk,
                            const float* __restrict__ bv,
                            half_t* __restrict__ Qh, half_t* __restrict__ Kh,
                            half_t* __restrict__ Vh) {
    int i = blockIdx.x * 256 + threadIdx.x;       // over TOK*1024
    int r = i >> 10, c = i & 1023;
    int h = c >> 6, k = c & 63;
    int b = r >> 11, s = r & 2047;
    size_t dst = ((size_t)(b * NHEADS + h) * SEQ + s) * DK + k;
    const float* row = QKV + (size_t)r * 3072;
    Qh[dst] = (half_t)((row[c]        + bq[h * DK + k]) * 0.125f);
    Kh[dst] = (half_t)( row[1024 + c] + bk[h * DK + k]);
    Vh[dst] = (half_t)( row[2048 + c] + bv[h * DK + k]);
}

// ---------------------------------------------------------------- flash attention
// Block = (b,h) x 128-query chunk; 8 waves, 16 query rows each; key tiles of 32.
// Double-buffered: next K tile DMA'd by TDM (TENSORcnt) and next V tile loaded to
// registers while current tile's WMMAs run; waits land just before the barrier.
__global__ void __launch_bounds__(256)
k_attention(const half_t* __restrict__ Qh, const half_t* __restrict__ Kh,
            const half_t* __restrict__ Vh, half_t* __restrict__ concatA) {
    __shared__ __align__(16) half_t Ks[2][32][72];    // K tiles [key][dim], stride 72
    __shared__ __align__(16) half_t Vs[2][64][40];    // V tiles transposed [dim][key]
    __shared__ __align__(16) half_t Ps[8][16][40];    // per-wave P staging [row][key]
    int tid = threadIdx.x, lane = tid & 31, wave = tid >> 5;
    int bh = blockIdx.y;                // b*16+h
    int b = bh >> 4, h = bh & 15;
    const half_t* Kbase = Kh + (size_t)bh * SEQ * DK;
    const half_t* Vbase = Vh + (size_t)bh * SEQ * DK;
    int qrow0 = blockIdx.x * 128 + wave * 16;
    int kb  = (lane >> 4) * 8;
    int l15 = lane & 15;

    Frag16 qa0, qa1;
    const half_t* qrow = Qh + ((size_t)bh * SEQ + qrow0 + l15) * DK;
    qa0.f4[0] = *(const float4*)(qrow + kb);
    qa0.f4[1] = *(const float4*)(qrow + kb + 16);
    qa1.f4[0] = *(const float4*)(qrow + 32 + kb);
    qa1.f4[1] = *(const float4*)(qrow + 32 + kb + 16);

    v8f accZ[4];
#pragma unroll
    for (int nt = 0; nt < 4; ++nt) accZ[nt] = zero8();
    float mrun[8], lrun[8];
#pragma unroll
    for (int j = 0; j < 8; ++j) { mrun[j] = -1e30f; lrun[j] = 0.0f; }

#if !HAVE_TDM
    int krow_l = tid >> 3, koff_l = (tid & 7) * 8;  // fallback K loader
#endif
    int vd = tid & 63, vkg = tid >> 6;              // V transposing loader
    unsigned int ks_lds[2] = { lds_addr_of(&Ks[0][0][0]), lds_addr_of(&Ks[1][0][0]) };

    // prologue: tile 0 into buffer 0
#if HAVE_TDM
    if (wave == 0) tdm_load_k_tile(Kbase, ks_lds[0]);
#else
    *(float4*)&Ks[0][krow_l][koff_l] =
        *(const float4*)(Kbase + (size_t)krow_l * DK + koff_l);
#endif
    {
        Half8 vt;
#pragma unroll
        for (int i = 0; i < 8; ++i)
            vt.h[i] = Vbase[(size_t)(vkg * 8 + i) * DK + vd];
        *(float4*)&Vs[0][vd][vkg * 8] = vt.f4;
    }
#if HAVE_TDM
    if (wave == 0) __builtin_amdgcn_s_wait_tensorcnt(0);
#endif
    __syncthreads();

    int buf = 0;
    for (int kt = 0; kt < SEQ; kt += 32) {
        int nxt = buf ^ 1;
        bool more = (kt + 32) < SEQ;
        Half8 vt;
        if (more) {
            // prefetch next tile: K via TDM, V into registers
#if HAVE_TDM
            if (wave == 0) tdm_load_k_tile(Kbase + (size_t)(kt + 32) * DK, ks_lds[nxt]);
#else
            *(float4*)&Ks[nxt][krow_l][koff_l] =
                *(const float4*)(Kbase + (size_t)(kt + 32 + krow_l) * DK + koff_l);
#endif
#pragma unroll
            for (int i = 0; i < 8; ++i)
                vt.h[i] = Vbase[(size_t)(kt + 32 + vkg * 8 + i) * DK + vd];
        }

        // scores: two 16-key groups, each = two chained WMMAs over the 64-dim reduction
        v8f accS[2];
#pragma unroll
        for (int g = 0; g < 2; ++g) {
            Frag16 bk0, bk1;
            int krow = g * 16 + l15;
            bk0.f4[0] = *(const float4*)&Ks[buf][krow][kb];
            bk0.f4[1] = *(const float4*)&Ks[buf][krow][kb + 16];
            bk1.f4[0] = *(const float4*)&Ks[buf][krow][32 + kb];
            bk1.f4[1] = *(const float4*)&Ks[buf][krow][32 + kb + 16];
            v8f z = zero8();
            z = __builtin_amdgcn_wmma_f32_16x16x32_f16(false, qa0.v, false, bk0.v, (short)0, z, false, false);
            z = __builtin_amdgcn_wmma_f32_16x16x32_f16(false, qa1.v, false, bk1.v, (short)0, z, false, false);
            accS[g] = z;
        }

        // online softmax; C layout: VGPR j -> row j (+8 for lanes 16-31), lane&15 -> col
#pragma unroll
        for (int j = 0; j < 8; ++j) {
            float s0 = accS[0][j], s1 = accS[1][j];
            float tm = fmaxf(s0, s1);
            for (int off = 1; off < 16; off <<= 1) tm = fmaxf(tm, __shfl_xor(tm, off, 32));
            float mnew  = fmaxf(mrun[j], tm);
            float scale = __expf(mrun[j] - mnew);
            float p0 = __expf(s0 - mnew);
            float p1 = __expf(s1 - mnew);
            float rs = p0 + p1;
            for (int off = 1; off < 16; off <<= 1) rs += __shfl_xor(rs, off, 32);
            lrun[j] = lrun[j] * scale + rs;
            mrun[j] = mnew;
#pragma unroll
            for (int nt = 0; nt < 4; ++nt) accZ[nt][j] *= scale;
            int prow = j + kb;                       // row j or j+8
            Ps[wave][prow][l15]      = (half_t)p0;
            Ps[wave][prow][l15 + 16] = (half_t)p1;
        }
        __syncthreads();                             // fence P staging; keep waves lockstep

        Frag16 pa;
        pa.f4[0] = *(const float4*)&Ps[wave][l15][kb];
        pa.f4[1] = *(const float4*)&Ps[wave][l15][kb + 16];
#pragma unroll
        for (int nt = 0; nt < 4; ++nt) {
            Frag16 bv;
            int dcol = nt * 16 + l15;
            bv.f4[0] = *(const float4*)&Vs[buf][dcol][kb];
            bv.f4[1] = *(const float4*)&Vs[buf][dcol][kb + 16];
            accZ[nt] = __builtin_amdgcn_wmma_f32_16x16x32_f16(
                false, pa.v, false, bv.v, (short)0, accZ[nt], false, false);
        }

        if (more) {
            *(float4*)&Vs[nxt][vd][vkg * 8] = vt.f4;
#if HAVE_TDM
            if (wave == 0) __builtin_amdgcn_s_wait_tensorcnt(0);
#endif
        }
        __syncthreads();
        buf = nxt;
    }

    // normalize and write to concat layout [b][s][h*64+k] as f16
#pragma unroll
    for (int j = 0; j < 8; ++j) {
        float inv = 1.0f / lrun[j];
        int srow = qrow0 + kb + j;
        size_t obase = ((size_t)b * SEQ + srow) * D_MODELC + h * DK;
#pragma unroll
        for (int nt = 0; nt < 4; ++nt)
            concatA[obase + nt * 16 + l15] = (half_t)(accZ[nt][j] * inv);
    }
}

// ---------------------------------------------------------------- residual + LayerNorm
__global__ void __launch_bounds__(256)
k_add_ln(const float* __restrict__ A, const float* __restrict__ Bv,
         const float* __restrict__ g, const float* __restrict__ be,
         float* __restrict__ out32, half_t* __restrict__ out16) {
    __shared__ float ssum[8], ssq[8];
    int r = blockIdx.x, tid = threadIdx.x;
    const float* a  = A  + (size_t)r * D_MODELC;
    const float* bb = Bv + (size_t)r * D_MODELC;
    float vals[4], s = 0.0f, q = 0.0f;
#pragma unroll
    for (int i = 0; i < 4; ++i) {
        int c = tid + i * 256;
        float v = a[c] + bb[c];
        vals[i] = v; s += v; q += v * v;
    }
    for (int off = 16; off; off >>= 1) { s += __shfl_xor(s, off, 32); q += __shfl_xor(q, off, 32); }
    int lane = tid & 31, wave = tid >> 5;
    if (lane == 0) { ssum[wave] = s; ssq[wave] = q; }
    __syncthreads();
    if (wave == 0) {
        s = ssum[lane & 7]; q = ssq[lane & 7];
        for (int off = 4; off; off >>= 1) { s += __shfl_xor(s, off, 32); q += __shfl_xor(q, off, 32); }
        if (lane == 0) { ssum[0] = s; ssq[0] = q; }
    }
    __syncthreads();
    s = ssum[0]; q = ssq[0];
    float mu   = s * (1.0f / D_MODELC);
    float var  = q * (1.0f / D_MODELC) - mu * mu;
    float rstd = rsqrtf(var + 1e-5f);
#pragma unroll
    for (int i = 0; i < 4; ++i) {
        int c = tid + i * 256;
        float v = (vals[i] - mu) * rstd * g[c] + be[c];
        if (out32) out32[(size_t)r * D_MODELC + c] = v;
        if (out16) out16[(size_t)r * D_MODELC + c] = (half_t)v;
    }
}

// ---------------------------------------------------------------- host
extern "C" void kernel_launch(void* const* d_in, const int* in_sizes, int n_in,
                              void* d_out, int out_size, void* d_ws, size_t ws_size,
                              hipStream_t stream) {
    const float* x   = (const float*)d_in[0];
    const float* Wq  = (const float*)d_in[1];
    const float* bq  = (const float*)d_in[2];
    const float* Wk  = (const float*)d_in[3];
    const float* bk  = (const float*)d_in[4];
    const float* Wv  = (const float*)d_in[5];
    const float* bv  = (const float*)d_in[6];
    const float* Wo  = (const float*)d_in[7];
    const float* bo  = (const float*)d_in[8];
    const float* W1  = (const float*)d_in[9];
    const float* b1  = (const float*)d_in[10];
    const float* W2  = (const float*)d_in[11];
    const float* b2  = (const float*)d_in[12];
    const float* g1  = (const float*)d_in[13];
    const float* be1 = (const float*)d_in[14];
    const float* g2  = (const float*)d_in[15];
    const float* be2 = (const float*)d_in[16];
    float* out = (float*)d_out;
    char* ws = (char*)d_ws;

    // lifetime-based workspace layout (peak ~184 MB)
    half_t* Wqkvh = (half_t*)(ws + 0);            // 6.0 MB   [persistent]
    half_t* Woh   = (half_t*)(ws + 6291456);      // 2.0 MB   [persistent]
    half_t* W1h   = (half_t*)(ws + 8388608);      // 8.0 MB   [persistent]
    half_t* W2h   = (half_t*)(ws + 16777216);     // 8.0 MB   [persistent]
    float*  QKVf  = (float*) (ws + 25165824);     // 96 MB    [ph2-ph3]
    half_t* xh    = (half_t*)(ws + 125829120);    // 16 MB    [ph1-ph2]
    half_t* Qh    = (half_t*)(ws + 125829120);    // 16 MB    [ph3-ph4] (reuses xh)
    half_t* Kh    = (half_t*)(ws + 142606336);    // 16 MB    [ph3-ph4]
    half_t* Vh    = (half_t*)(ws + 159383552);    // 16 MB    [ph3-ph4]
    half_t* catA  = (half_t*)(ws + 25165824);     // 16 MB    [ph4-ph5] (reuses QKVf)
    float*  attno = (float*) (ws + 41943040);     // 32 MB    [ph5-ph6]
    float*  hf32  = (float*) (ws + 75497472);     // 32 MB    [ph6-ph9]
    half_t* hf16  = (half_t*)(ws + 109051904);    // 16 MB    [ph6-ph7]
    half_t* ffn1h = (half_t*)(ws + 125829120);    // 64 MB    [ph7-ph8] (reuses Q/K/Vh)
    float*  ffn2f = (float*) (ws + 25165824);     // 32 MB    [ph8-ph9] (reuses catA)

    // phase 1: conversions
    k_f32_to_f16<<<(TOK * D_MODELC) / 256, 256, 0, stream>>>(x, xh, TOK * D_MODELC);
    k_f32_to_f16<<<(D_MODELC * D_MODELC) / 256, 256, 0, stream>>>(Wo, Woh, D_MODELC * D_MODELC);
    k_f32_to_f16<<<(D_MODELC * FFNC) / 256, 256, 0, stream>>>(W1, W1h, D_MODELC * FFNC);
    k_f32_to_f16<<<(FFNC * D_MODELC) / 256, 256, 0, stream>>>(W2, W2h, FFNC * D_MODELC);
    k_build_wqkv<<<(D_MODELC * 3072) / 256, 256, 0, stream>>>(Wq, Wk, Wv, Wqkvh);

    // phase 2: fused QKV projection GEMM  (8192 x 3072 x 1024)
    k_gemm_f16<false, false, true, false><<<dim3(3072 / 128, TOK / 128), 256, 0, stream>>>(
        xh, Wqkvh, nullptr, QKVf, nullptr, TOK, 3072, 1024);

    // phase 3: bias + head-major split (Q prescaled by 1/8)
    k_qkv_split<<<(TOK * D_MODELC) / 256, 256, 0, stream>>>(QKVf, bq, bk, bv, Qh, Kh, Vh);

    // phase 4: flash attention -> concat layout f16
    k_attention<<<dim3(SEQ / 128, BATCH * NHEADS), 256, 0, stream>>>(Qh, Kh, Vh, catA);

    // phase 5: output projection  (8192 x 1024 x 1024) + bo
    k_gemm_f16<true, false, true, false><<<dim3(D_MODELC / 128, TOK / 128), 256, 0, stream>>>(
        catA, Woh, bo, attno, nullptr, TOK, D_MODELC, D_MODELC);

    // phase 6: h = LN(x + attn_out)
    k_add_ln<<<TOK, 256, 0, stream>>>(x, attno, g1, be1, hf32, hf16);

    // phase 7: FFN1 (8192 x 4096 x 1024) + b1 + ReLU -> f16 only
    k_gemm_f16<true, true, false, true><<<dim3(FFNC / 128, TOK / 128), 256, 0, stream>>>(
        hf16, W1h, b1, nullptr, ffn1h, TOK, FFNC, D_MODELC);

    // phase 8: FFN2 (8192 x 1024 x 4096) + b2
    k_gemm_f16<true, false, true, false><<<dim3(D_MODELC / 128, TOK / 128), 256, 0, stream>>>(
        ffn1h, W2h, b2, ffn2f, nullptr, TOK, D_MODELC, FFNC);

    // phase 9: out = LN(h + ffn_out)
    k_add_ln<<<TOK, 256, 0, stream>>>(hf32, ffn2f, g2, be2, out, nullptr);
}
